// MultiHeadAttention_51221779972138
// MI455X (gfx1250) — compile-verified, matrix operand order checked by reading, not run on hardware
//
#include <hip/hip_runtime.h>

#define D_MODEL   2048
#define NUM_HEADS 16
#define HEAD_DIM  128
#define BATCH     2
#define SEQ       2048

typedef __bf16          bf16x16 __attribute__((ext_vector_type(16)));
typedef float           f32x8   __attribute__((ext_vector_type(8)));
typedef unsigned short  us8     __attribute__((ext_vector_type(8)));
typedef unsigned short  us4     __attribute__((ext_vector_type(4)));

union FragBF { bf16x16 v; us8 u[2]; };

__device__ __forceinline__ unsigned short f2bf(float f) {
  unsigned int u = __builtin_bit_cast(unsigned int, f);
  u += 0x7fffu + ((u >> 16) & 1u);            // round-to-nearest-even
  return (unsigned short)(u >> 16);
}

__device__ __forceinline__ f32x8 wmma_bf16(const FragBF& a, const FragBF& b, f32x8 c) {
  return __builtin_amdgcn_wmma_f32_16x16x32_bf16(false, a.v, false, b.v,
                                                 (short)0, c, false, false);
}

// Async global->LDS copy, 16B per lane, tracked by ASYNCcnt.
// lds_off: byte offset within LDS (low 32 bits of a generic LDS pointer).
__device__ __forceinline__ void async_ld16(unsigned lds_off, const void* gptr) {
  asm volatile("global_load_async_to_lds_b128 %0, %1, off"
               :: "v"(lds_off), "v"(gptr) : "memory");
}
__device__ __forceinline__ void wait_async0() {
  asm volatile("s_wait_asynccnt 0" ::: "memory");
}

// ---------------------------------------------------------------- fp32 -> bf16
__global__ __launch_bounds__(256)
void cvt_f32_bf16(const float* __restrict__ in, unsigned short* __restrict__ out, int n) {
  int i = (blockIdx.x * 256 + threadIdx.x) * 4;
  if (i + 3 < n) {
    float4 f = *reinterpret_cast<const float4*>(in + i);
    us4 o;
    o[0] = f2bf(f.x); o[1] = f2bf(f.y); o[2] = f2bf(f.z); o[3] = f2bf(f.w);
    *reinterpret_cast<us4*>(out + i) = o;
  }
}

// ------------------------------------------------- C[M,N] = A[M,K] * B[N,K]^T
// bf16 in, f32 accumulate. Block tile 128x128, 8 waves, wave tile 64x32.
// Double-buffered LDS fed by async global->LDS copies (ASYNCcnt pipeline).
template<int OUT_BF16>
__global__ __launch_bounds__(256)
void gemm_nt(const unsigned short* __restrict__ A,
             const unsigned short* __restrict__ B,
             void* __restrict__ Cout, int M, int N, int K)
{
  __shared__ __align__(16) unsigned short As[2][128][40];  // padded: 80B rows
  __shared__ __align__(16) unsigned short Bs[2][128][40];

  const int tid  = threadIdx.x;
  const int wave = tid >> 5, lane = tid & 31;
  const int r16  = lane & 15, hi = lane >> 4, ko = hi * 8;
  const int wm   = wave >> 2, wn = wave & 3;
  const int bm   = blockIdx.y * 128, bn = blockIdx.x * 128;

  auto stage = [&](int k0, int buf) {
#pragma unroll
    for (int i = 0; i < 2; ++i) {
      int c = tid + i * 256;                   // 512 chunks of 8 bf16 per tile
      int row = c >> 2, ch = (c & 3) * 8;
      async_ld16((unsigned)(uintptr_t)&As[buf][row][ch],
                 &A[(size_t)(bm + row) * K + k0 + ch]);
      async_ld16((unsigned)(uintptr_t)&Bs[buf][row][ch],
                 &B[(size_t)(bn + row) * K + k0 + ch]);
    }
  };

  const f32x8 zero = {0.f,0.f,0.f,0.f,0.f,0.f,0.f,0.f};
  f32x8 acc[4][2];
#pragma unroll
  for (int f = 0; f < 4; ++f)
#pragma unroll
    for (int g = 0; g < 2; ++g) acc[f][g] = zero;

  stage(0, 0);                                 // prime the pipeline

  for (int k0 = 0; k0 < K; k0 += 32) {
    const int cur = (k0 >> 5) & 1;
    wait_async0();                             // this wave's copies landed
    __syncthreads();                           // everyone's copies landed
    if (k0 + 32 < K) stage(k0 + 32, cur ^ 1);  // stream next tile under compute
    if (k0 + 64 < K) {                         // L2 warm-up two tiles ahead
      __builtin_prefetch(&A[(size_t)(bm + (tid >> 1)) * K + k0 + 64], 0, 0);
      __builtin_prefetch(&B[(size_t)(bn + (tid >> 1)) * K + k0 + 64], 0, 0);
    }

    FragBF af[4], bfr[2];
#pragma unroll
    for (int f = 0; f < 4; ++f) {
      const int row = wm * 64 + f * 16 + r16;
      af[f].u[0] = *reinterpret_cast<const us8*>(&As[cur][row][ko]);
      af[f].u[1] = *reinterpret_cast<const us8*>(&As[cur][row][ko + 16]);
    }
#pragma unroll
    for (int g = 0; g < 2; ++g) {
      const int row = wn * 32 + g * 16 + r16;
      bfr[g].u[0] = *reinterpret_cast<const us8*>(&Bs[cur][row][ko]);
      bfr[g].u[1] = *reinterpret_cast<const us8*>(&Bs[cur][row][ko + 16]);
    }
#pragma unroll
    for (int f = 0; f < 4; ++f)
#pragma unroll
      for (int g = 0; g < 2; ++g)
        acc[f][g] = wmma_bf16(af[f], bfr[g], acc[f][g]);
  }

#pragma unroll
  for (int f = 0; f < 4; ++f)
#pragma unroll
    for (int g = 0; g < 2; ++g)
#pragma unroll
      for (int v = 0; v < 8; ++v) {
        int row = bm + wm * 64 + f * 16 + v + 8 * hi;
        int col = bn + wn * 32 + g * 16 + r16;
        float val = acc[f][g][v];
        if (OUT_BF16)
          ((unsigned short*)Cout)[(size_t)row * N + col] = f2bf(val);
        else
          ((float*)Cout)[(size_t)row * N + col] = val;
      }
}

// --------------------------------------------- causal flash attention (bf16)
// Grid: (SEQ/128, NUM_HEADS, BATCH). Block: 256 threads (8 waves).
// Wave owns 16 query rows; K/V streamed 32 keys at a time through LDS.
// K tile staged via async global->LDS; V transposed through the VGPR path.
__global__ __launch_bounds__(256)
void attn_kernel(const unsigned short* __restrict__ Qb,
                 const unsigned short* __restrict__ Kb,
                 const unsigned short* __restrict__ Vb,
                 unsigned short* __restrict__ ctx)
{
  __shared__ __align__(16) unsigned short Ks[32][136];   // [key][d], padded
  __shared__ __align__(16) unsigned short Vt[128][40];   // [d][key], padded
  __shared__ __align__(16) unsigned short Ps[8][16][40]; // per-wave P scratch

  const int tid  = threadIdx.x;
  const int wave = tid >> 5, lane = tid & 31;
  const int r16  = lane & 15, hi = lane >> 4, ko = hi * 8;
  const int qt   = blockIdx.x * 128;
  const int h    = blockIdx.y, b = blockIdx.z;

  // "plain view" head split: head (b,h) = flat offset b*S*D + h*S*Dh
  const size_t headoff = (size_t)b * SEQ * D_MODEL + (size_t)h * SEQ * HEAD_DIM;
  const unsigned short* Qh = Qb + headoff;
  const unsigned short* Kh = Kb + headoff;
  const unsigned short* Vh = Vb + headoff;

  const int qbase = qt + wave * 16;

  FragBF qf[4];                                   // Q rows resident in regs
#pragma unroll
  for (int f = 0; f < 4; ++f) {
    const unsigned short* qp = &Qh[(size_t)(qbase + r16) * HEAD_DIM + f * 32 + ko];
    qf[f].u[0] = *reinterpret_cast<const us8*>(qp);
    qf[f].u[1] = *reinterpret_cast<const us8*>(qp + 16);
  }

  const f32x8 zero = {0.f,0.f,0.f,0.f,0.f,0.f,0.f,0.f};
  f32x8 oacc[8];
#pragma unroll
  for (int g = 0; g < 8; ++g) oacc[g] = zero;

  float m_run[8], l_run[8];
#pragma unroll
  for (int v = 0; v < 8; ++v) { m_run[v] = -1e30f; l_run[v] = 0.f; }

  const int jend = qt + 128;                      // causal upper bound
  for (int j = 0; j < jend; j += 32) {
    __syncthreads();
#pragma unroll
    for (int i = 0; i < 2; ++i) {                 // stage K async; V transposed
      int c = tid + i * 256;
      int row = c >> 4, ch = (c & 15) * 8;
      async_ld16((unsigned)(uintptr_t)&Ks[row][ch],
                 &Kh[(size_t)(j + row) * HEAD_DIM + ch]);
      us8 vv = *reinterpret_cast<const us8*>(&Vh[(size_t)(j + row) * HEAD_DIM + ch]);
#pragma unroll
      for (int e = 0; e < 8; ++e) Vt[ch + e][row] = vv[e];
    }
    wait_async0();
    __syncthreads();

    f32x8 s[2]; s[0] = zero; s[1] = zero;         // S = Q * K^T (K-dim = Dh)
#pragma unroll
    for (int kt = 0; kt < 2; ++kt)
#pragma unroll
      for (int f = 0; f < 4; ++f) {
        FragBF kf;
        const unsigned short* kp = &Ks[kt * 16 + r16][f * 32 + ko];
        kf.u[0] = *reinterpret_cast<const us8*>(kp);
        kf.u[1] = *reinterpret_cast<const us8*>(kp + 16);
        s[kt] = wmma_bf16(qf[f], kf, s[kt]);
      }

    float s0[8], s1[8], p0[8], p1[8], alpha[8];
#pragma unroll
    for (int v = 0; v < 8; ++v) {
      const int q   = qbase + v + 8 * hi;
      const int k0i = j + r16;
      const int k1i = j + 16 + r16;
      s0[v] = (k0i > q) ? -1e30f : s[0][v] * (1.0f / 128.0f);  // scale by 1/Dh
      s1[v] = (k1i > q) ? -1e30f : s[1][v] * (1.0f / 128.0f);
      float t = fmaxf(s0[v], s1[v]);
#pragma unroll
      for (int mm = 1; mm < 16; mm <<= 1)         // half-wave row reduction
        t = fmaxf(t, __shfl_xor(t, mm, 32));
      const float m_new = fmaxf(m_run[v], t);
      alpha[v] = __expf(m_run[v] - m_new);
      p0[v] = __expf(s0[v] - m_new);
      p1[v] = __expf(s1[v] - m_new);
      float rs = p0[v] + p1[v];
#pragma unroll
      for (int mm = 1; mm < 16; mm <<= 1)
        rs += __shfl_xor(rs, mm, 32);
      l_run[v] = l_run[v] * alpha[v] + rs;
      m_run[v] = m_new;
    }

#pragma unroll
    for (int g = 0; g < 8; ++g)                   // rescale running output
#pragma unroll
      for (int v = 0; v < 8; ++v)
        oacc[g][v] *= alpha[v];

#pragma unroll
    for (int v = 0; v < 8; ++v) {                 // relayout P via LDS
      Ps[wave][v + 8 * hi][r16]      = f2bf(p0[v]);
      Ps[wave][v + 8 * hi][16 + r16] = f2bf(p1[v]);
    }
    asm volatile("s_wait_dscnt 0" ::: "memory");  // wave-local DS fence

    FragBF pf;
    pf.u[0] = *reinterpret_cast<const us8*>(&Ps[wave][r16][ko]);
    pf.u[1] = *reinterpret_cast<const us8*>(&Ps[wave][r16][ko + 16]);

#pragma unroll
    for (int g = 0; g < 8; ++g) {                 // O += P * V (K-dim = keys)
      FragBF vf;
      const unsigned short* vp = &Vt[g * 16 + r16][ko];
      vf.u[0] = *reinterpret_cast<const us8*>(vp);
      vf.u[1] = *reinterpret_cast<const us8*>(vp + 16);
      oacc[g] = wmma_bf16(pf, vf, oacc[g]);
    }
  }

  // normalize; merge heads: ctx[b*S + q][h*Dh + d]
#pragma unroll
  for (int v = 0; v < 8; ++v) {
    const float inv = 1.0f / l_run[v];
    const int q = qbase + v + 8 * hi;
    unsigned short* crow = ctx + ((size_t)(b * SEQ + q) * D_MODEL) + h * HEAD_DIM;
#pragma unroll
    for (int g = 0; g < 8; ++g)
      crow[g * 16 + r16] = f2bf(oacc[g][v] * inv);
  }
}

// ---------------------------------------------------------------------- host
extern "C" void kernel_launch(void* const* d_in, const int* in_sizes, int n_in,
                              void* d_out, int out_size, void* d_ws, size_t ws_size,
                              hipStream_t stream) {
  (void)in_sizes; (void)n_in; (void)out_size; (void)ws_size;
  const float* x  = (const float*)d_in[0];
  const float* Wq = (const float*)d_in[1];
  const float* Wk = (const float*)d_in[2];
  const float* Wv = (const float*)d_in[3];
  const float* Wo = (const float*)d_in[4];

  const int M = BATCH * SEQ;               // 4096
  const int D = D_MODEL;                   // 2048
  const size_t szX = (size_t)M * D;        // 8,388,608 elems
  const size_t szW = (size_t)D * D;        // 4,194,304 elems

  unsigned short* ws  = (unsigned short*)d_ws;   // total 117,440,512 bytes
  unsigned short* xb  = ws;
  unsigned short* wqb = xb  + szX;
  unsigned short* wkb = wqb + szW;
  unsigned short* wvb = wkb + szW;
  unsigned short* wob = wvb + szW;
  unsigned short* qb  = wob + szW;
  unsigned short* kb  = qb  + szX;
  unsigned short* vb  = kb  + szX;
  unsigned short* cb  = vb  + szX;

  dim3 blk(256);
  cvt_f32_bf16<<<(unsigned)(szX / 4 / 256), blk, 0, stream>>>(x,  xb,  (int)szX);
  cvt_f32_bf16<<<(unsigned)(szW / 4 / 256), blk, 0, stream>>>(Wq, wqb, (int)szW);
  cvt_f32_bf16<<<(unsigned)(szW / 4 / 256), blk, 0, stream>>>(Wk, wkb, (int)szW);
  cvt_f32_bf16<<<(unsigned)(szW / 4 / 256), blk, 0, stream>>>(Wv, wvb, (int)szW);
  cvt_f32_bf16<<<(unsigned)(szW / 4 / 256), blk, 0, stream>>>(Wo, wob, (int)szW);

  dim3 ggrid(D / 128, M / 128);            // (16, 32)
  gemm_nt<1><<<ggrid, blk, 0, stream>>>(xb, wqb, qb, M, D, D);
  gemm_nt<1><<<ggrid, blk, 0, stream>>>(xb, wkb, kb, M, D, D);
  gemm_nt<1><<<ggrid, blk, 0, stream>>>(xb, wvb, vb, M, D, D);

  dim3 agrid(SEQ / 128, NUM_HEADS, BATCH); // (16, 16, 2)
  attn_kernel<<<agrid, blk, 0, stream>>>(qb, kb, vb, cb);

  gemm_nt<0><<<ggrid, blk, 0, stream>>>(cb, wob, d_out, M, D, D);
}
